// GumbelGraphNetwork_49469433315863
// MI455X (gfx1250) — matrix operand order repaired and checked by reading, not compiled, not run on hardware
//
#include <hip/hip_runtime.h>
#include <hip/hip_bf16.h>

typedef __attribute__((ext_vector_type(16))) _Float16 v16h;
typedef __attribute__((ext_vector_type(8)))  _Float16 v8h;
typedef __attribute__((ext_vector_type(8)))  float    v8f;

#define B_     4
#define N_     384
#define IN_    64
#define H_     256
#define OUTC   128      // OUT*IN
#define NT     24       // N/16
#define JSPLIT 4
#define JT_PER 6        // NT / JSPLIT

// ---------------------------------------------------------------------------
// Stage 0a: pi = x @ Wi^T + b_e1  (f16),  pj = x @ Wj^T  (f16)
// Wj = W_e1[:, :64], Wi = W_e1[:, 64:]
// ---------------------------------------------------------------------------
__global__ __launch_bounds__(256)
void k_prep_pipj(const float* __restrict__ x, const float* __restrict__ W_e1,
                 const float* __restrict__ b_e1,
                 _Float16* __restrict__ pi_h, _Float16* __restrict__ pj_h) {
  int idx = blockIdx.x * 256 + threadIdx.x;     // (b*N + n)*H + h
  int h  = idx & (H_ - 1);
  int bn = idx >> 8;
  const float* xr = x + bn * IN_;
  const float* wj = W_e1 + h * (2 * IN_);
  const float* wi = wj + IN_;
  float aj = 0.f, ai = 0.f;
#pragma unroll 16
  for (int d = 0; d < IN_; ++d) { float xv = xr[d]; aj += xv * wj[d]; ai += xv * wi[d]; }
  pi_h[idx] = (_Float16)(ai + b_e1[h]);
  pj_h[idx] = (_Float16)aj;
}

// ---------------------------------------------------------------------------
// Stage 0b: pre-swizzle W_e2 into per-lane WMMA B-fragment layout (f16)
// bsw[((gt*8+kt)*32+lane)*16+s] = W_e2[g, h],  g=gt*16+(lane&15),
// h = kt*32 + ((lane<16)?0:8) + ((s<8)?s:s+8)
// ---------------------------------------------------------------------------
__global__ __launch_bounds__(256)
void k_prep_bsw(const float* __restrict__ W_e2, _Float16* __restrict__ bsw) {
  int idx  = blockIdx.x * 256 + threadIdx.x;    // 65536 total
  int s    = idx & 15;
  int lane = (idx >> 4) & 31;
  int kt   = (idx >> 9) & 7;
  int gt   = idx >> 12;
  int g    = gt * 16 + (lane & 15);
  int h    = kt * 32 + ((lane < 16) ? 0 : 8) + ((s < 8) ? s : s + 8);
  bsw[idx] = (_Float16)W_e2[g * H_ + h];
}

// ---------------------------------------------------------------------------
// Stage 1: fused  relu(pi+pj) @ W_e2^T -> +b_e2 -> relu -> adj-weighted sum_j
// One WG = (batch b, 16-row i tile, 1/4 of the j range). 8 waves, each owns
// two 16-wide g tiles. All 16 B fragments cached in registers for the WG.
// ---------------------------------------------------------------------------
__global__ __launch_bounds__(256, 1)
void k_edge_gemm(const _Float16* __restrict__ pi_h, const _Float16* __restrict__ pj_h,
                 const _Float16* __restrict__ bsw, const float* __restrict__ adj,
                 const float* __restrict__ b_e2, float* __restrict__ partial) {
  __shared__ _Float16 piT[16][264];   // +8 halves pad -> conflict-free b128 reads
  __shared__ _Float16 pjT[16][264];
  __shared__ float    adjT[16][16];
  __shared__ float    aggL[16][256];

  const int tid = threadIdx.x;
  const int w   = tid >> 5;
  const int l   = tid & 31;
  const int jc = blockIdx.x, it = blockIdx.y, b = blockIdx.z;
  const int ibase = it * 16;

#pragma unroll
  for (int r = 0; r < 16; ++r) aggL[r][tid] = 0.f;

  { // stage pi tile (16 x 256 f16)
    int il = tid >> 4, ch = tid & 15;
    const v8h* src = (const v8h*)(pi_h + ((b * N_ + ibase + il) * H_ + ch * 16));
    *(v8h*)&piT[il][ch * 16]     = src[0];
    *(v8h*)&piT[il][ch * 16 + 8] = src[1];
  }

  // cache all B fragments in registers: 2 g-tiles x 8 k-steps per wave
  v16h breg[16];
#pragma unroll
  for (int f = 0; f < 16; ++f) {
    int gt = 2 * w + (f >> 3);
    int kt = f & 7;
    breg[f] = *(const v16h*)(bsw + (((gt * 8 + kt) * 32 + l) * 16));
  }

  const int n     = l & 15;
  const int rbase = (l < 16) ? 0 : 8;   // D-tile rows this lane owns
  const int kb    = (l < 16) ? 0 : 8;   // A-fragment K sub-block
  const int jrow  = l & 15;             // A-fragment row (j_local)
  const int g0    = w * 32 + n;
  const int g1    = g0 + 16;
  const float bias0 = b_e2[g0];
  const float bias1 = b_e2[g1];

  for (int jj = 0; jj < JT_PER; ++jj) {
    const int jbase = (jc * JT_PER + jj) * 16;
    __syncthreads();
    { // stage pj tile + adj tile
      int jl = tid >> 4, ch = tid & 15;
      const v8h* src = (const v8h*)(pj_h + ((b * N_ + jbase + jl) * H_ + ch * 16));
      *(v8h*)&pjT[jl][ch * 16]     = src[0];
      *(v8h*)&pjT[jl][ch * 16 + 8] = src[1];
      adjT[tid >> 4][tid & 15] =
          adj[(b * N_ + ibase + (tid >> 4)) * N_ + jbase + (tid & 15)];
    }
    __syncthreads();

    for (int t = 0; t < 16; ++t) {      // i_local = M-tile index
      float adjv[8];
#pragma unroll
      for (int v = 0; v < 8; ++v) adjv[v] = adjT[t][rbase + v];
      v8f acc0 = {}; v8f acc1 = {};
#pragma unroll
      for (int kt = 0; kt < 8; ++kt) {
        const int h0 = kt * 32 + kb;    // slots 0..7  -> K kb..kb+7
        const int h1 = h0 + 16;         // slots 8..15 -> K 16+kb..
        v8h s0 = *(const v8h*)&piT[t][h0] + *(const v8h*)&pjT[jrow][h0];
        v8h s1 = *(const v8h*)&piT[t][h1] + *(const v8h*)&pjT[jrow][h1];
#pragma unroll
        for (int s = 0; s < 8; ++s) {
          s0[s] = (s0[s] > (_Float16)0) ? s0[s] : (_Float16)0;
          s1[s] = (s1[s] > (_Float16)0) ? s1[s] : (_Float16)0;
        }
        v16h a;
#pragma unroll
        for (int s = 0; s < 8; ++s) { a[s] = s0[s]; a[s + 8] = s1[s]; }
        acc0 = __builtin_amdgcn_wmma_f32_16x16x32_f16(false, a, false, breg[kt],
                                                      (short)0, acc0, false, false);
        acc1 = __builtin_amdgcn_wmma_f32_16x16x32_f16(false, a, false, breg[kt + 8],
                                                      (short)0, acc1, false, false);
      }
      // epilogue: +b_e2, relu, adj-weighted row reduction into LDS accumulator
      float r0 = 0.f, r1 = 0.f;
#pragma unroll
      for (int v = 0; v < 8; ++v) {
        float e0 = fmaxf(acc0[v] + bias0, 0.f);
        float e1 = fmaxf(acc1[v] + bias1, 0.f);
        r0 += adjv[v] * e0;
        r1 += adjv[v] * e1;
      }
      atomicAdd(&aggL[t][g0], r0);
      atomicAdd(&aggL[t][g1], r1);
    }
  }
  __syncthreads();
  float* dst = partial + ((((b * NT + it) * JSPLIT) + jc) * 16 * 256);
#pragma unroll
  for (int r = 0; r < 16; ++r) dst[r * 256 + tid] = aggL[r][tid];
}

// ---------------------------------------------------------------------------
// Stage 2: reduce j-chunk partials; node MLPs; output head. One WG per (b,i).
// ---------------------------------------------------------------------------
__global__ __launch_bounds__(256)
void k_tail(const float* __restrict__ partial, const float* __restrict__ x,
            const float* __restrict__ W_n1, const float* __restrict__ b_n1,
            const float* __restrict__ W_n2, const float* __restrict__ b_n2,
            const float* __restrict__ W_out, const float* __restrict__ b_out,
            float* __restrict__ out) {
  __shared__ float row[256];
  __shared__ float mid[256];
  const int tid = threadIdx.x;
  const int bi  = blockIdx.x;         // b*N + i
  const int b   = bi / N_;
  const int i   = bi - b * N_;
  const int it = i >> 4, il = i & 15;
  float s = 0.f;
#pragma unroll
  for (int jc = 0; jc < JSPLIT; ++jc)
    s += partial[(((b * NT + it) * JSPLIT + jc) * 16 + il) * 256 + tid];
  row[tid] = s;
  __syncthreads();
  {
    const float* wr = W_n1 + tid * H_;
    float a = b_n1[tid];
#pragma unroll 8
    for (int h = 0; h < H_; ++h) a += row[h] * wr[h];
    mid[tid] = fmaxf(a, 0.f);
  }
  __syncthreads();
  {
    const float* wr = W_n2 + tid * H_;
    float a = b_n2[tid];
#pragma unroll 8
    for (int h = 0; h < H_; ++h) a += mid[h] * wr[h];
    row[tid] = fmaxf(a, 0.f);
  }
  __syncthreads();
  if (tid < OUTC) {
    const float* wr = W_out + tid * (IN_ + H_);
    const float* xr = x + bi * IN_;
    float a = b_out[tid];
#pragma unroll 8
    for (int d = 0; d < IN_; ++d) a += xr[d] * wr[d];
#pragma unroll 8
    for (int h = 0; h < H_; ++h) a += row[h] * wr[IN_ + h];
    out[bi * OUTC + tid] = a;
  }
}

// ---------------------------------------------------------------------------
extern "C" void kernel_launch(void* const* d_in, const int* in_sizes, int n_in,
                              void* d_out, int out_size, void* d_ws, size_t ws_size,
                              hipStream_t stream) {
  const float* x     = (const float*)d_in[0];
  const float* adj   = (const float*)d_in[1];
  const float* W_e1  = (const float*)d_in[2];
  const float* b_e1  = (const float*)d_in[3];
  const float* W_e2  = (const float*)d_in[4];
  const float* b_e2  = (const float*)d_in[5];
  const float* W_n1  = (const float*)d_in[6];
  const float* b_n1  = (const float*)d_in[7];
  const float* W_n2  = (const float*)d_in[8];
  const float* b_n2  = (const float*)d_in[9];
  const float* W_out = (const float*)d_in[10];
  const float* b_out = (const float*)d_in[11];
  float* out = (float*)d_out;

  uint8_t* ws = (uint8_t*)d_ws;
  _Float16* pi_h = (_Float16*)(ws);                 // 768 KB
  _Float16* pj_h = (_Float16*)(ws + 786432);        // 768 KB
  _Float16* bsw  = (_Float16*)(ws + 1572864);       // 128 KB
  float*  partial = (float*)(ws + 1703936);         // 6 MB

  k_prep_pipj<<<(B_ * N_ * H_) / 256, 256, 0, stream>>>(x, W_e1, b_e1, pi_h, pj_h);
  k_prep_bsw<<<65536 / 256, 256, 0, stream>>>(W_e2, bsw);
  dim3 g(JSPLIT, NT, B_);
  k_edge_gemm<<<g, 256, 0, stream>>>(pi_h, pj_h, bsw, adj, b_e2, partial);
  k_tail<<<B_ * N_, 256, 0, stream>>>(partial, x, W_n1, b_n1, W_n2, b_n2,
                                      W_out, b_out, out);
}
